// MOPN_Layer_80599356277464
// MI455X (gfx1250) — matrix operand order repaired and checked by reading, not compiled, not run on hardware
//
#include <hip/hip_runtime.h>

// ---------------------------------------------------------------------------
// MOPN layer for MI455X (gfx1250, wave32, WMMA).
//   x:[4,4096,1024] f32, norm_w:[1024], Wg/Wu:[4,256,512], Wd:[4,512,256],
//   Wo:[1024,1024].  out = (grouped-SwiGLU(RoPE(RMSNorm(x)))) @ Wo + x
// Matmuls: bf16 WMMA, f32 accumulate. Norm/RoPE/SiLU/residual: f32.
// M-tile = 32 tokens; each wave computes 2 M-subtiles so every B fragment
// feeds 2 WMMAs (halves L2 weight traffic vs 16-token tiles).
// ---------------------------------------------------------------------------

typedef __bf16 bf16;
typedef __attribute__((ext_vector_type(16))) __bf16 v16bf;
typedef __attribute__((ext_vector_type(8)))  float  v8f;

constexpr int kBn = 4, kSn = 4096, kDn = 1024;
constexpr int kNS = 4, kDS = 256, kHn = 512;
constexpr int kT  = kBn * kSn;      // 16384 tokens
constexpr int kMT = 32;             // tokens per workgroup (2 WMMA M tiles)

// ws layout (bytes); total ~38.8 MB
constexpr size_t kDnOff  = 0;
constexpr size_t kDnSz   = (size_t)kT * kDn * 2;                 // 32 MB bf16
constexpr size_t kWgTOff = kDnOff + kDnSz;
constexpr size_t kWTSz   = (size_t)kNS * kHn * kDS * 2;          // 1 MB
constexpr size_t kWuTOff = kWgTOff + kWTSz;
constexpr size_t kWdTOff = kWuTOff + kWTSz;
constexpr size_t kWoTOff = kWdTOff + kWTSz;                      // +2 MB

union Frag16 { v16bf v; uint4 q[2]; };

// A-operand fragment: two 16B chunks (K 0..15 for lanes 0-15, K 16..31 for
// lanes 16-31), contiguous K order matching the B fragment.
__device__ __forceinline__ v16bf load_a2(const bf16* c0, const bf16* c1) {
  Frag16 f;
  f.q[0] = *(const uint4*)c0;
  f.q[1] = *(const uint4*)c1;
  return f.v;
}
// B-operand fragment: 16 contiguous K values of this lane's column.
__device__ __forceinline__ v16bf load_b16(const bf16* p) {
  Frag16 f;
  const uint4* q = (const uint4*)p;
  f.q[0] = q[0];
  f.q[1] = q[1];
  return f.v;
}

__device__ __forceinline__ v8f wmma_bf16(v16bf a, v16bf b, v8f c) {
  return __builtin_amdgcn_wmma_f32_16x16x32_bf16(
      /*neg_a=*/false, a, /*neg_b=*/false, b,
      /*c_mod=*/(short)0, c, /*reuse_a=*/false, /*reuse_b=*/false);
}

// ---------------------------------------------------------------------------
// Weight prep: dst[c*R + r] = (bf16)src[r*C + c]   (transpose + convert)
// ---------------------------------------------------------------------------
__global__ void transpose_cvt(const float* __restrict__ src,
                              bf16* __restrict__ dst, int R, int C) {
  int i = blockIdx.x * blockDim.x + threadIdx.x;
  if (i >= R * C) return;
  int r = i / C, c = i % C;
  dst[(size_t)c * R + r] = (bf16)src[i];
}

// ---------------------------------------------------------------------------
// Fused RMSNorm + RoPE + grouped SwiGLU.  One block = 32 tokens, 8 waves.
// Per subspace: re-materialize the RoPE'd 32x256 hidden slice into LDS,
// gate/up GEMMs (K=256), SiLU*up -> LDS, down GEMM (K=512) -> bf16 dn.
// ---------------------------------------------------------------------------
__global__ __launch_bounds__(256) void fused_norm_rope_glu(
    const float* __restrict__ x, const float* __restrict__ norm_w,
    bf16* __restrict__ dn, const bf16* __restrict__ WgT,
    const bf16* __restrict__ WuT, const bf16* __restrict__ WdT) {
  __shared__ __align__(16) bf16 hS[kMT][kDS + 8];   // subspace slice, bf16
  __shared__ __align__(16) bf16 sS[kMT][kHn + 8];   // silu(gate)*up, bf16
  __shared__ float rmsS[kMT];

  const int tid  = threadIdx.x;
  const int lane = tid & 31;
  const int wave = tid >> 5;
  const int t0   = blockIdx.x * kMT;

  // ---- Phase 0: per-row RMS (f32), 4 rows per wave ----
  for (int rr = 0; rr < 4; ++rr) {
    const int r = wave * 4 + rr;
    const float* xr = x + (size_t)(t0 + r) * kDn;
    float ss = 0.f;
    for (int d = lane; d < kDn; d += 32) { float v = xr[d]; ss += v * v; }
    #pragma unroll
    for (int off = 16; off; off >>= 1) ss += __shfl_xor(ss, off, 32);
    if (lane == 0) rmsS[r] = rsqrtf(ss * (1.0f / kDn) + 1e-6f);
  }
  __syncthreads();

  const int rowA = lane & 15;          // A fragment row (M within tile)
  const int hiA  = lane >> 4;          // half-wave selector
  const int colB = lane & 15;          // B/C column (N within tile)
  const int kHi  = hiA << 4;           // B fragment K offset (0 or 16)

  for (int n = 0; n < kNS; ++n) {
    // ---- stage RoPE'd slice hc[:, n*256 : n*256+256] as bf16 ----
    __syncthreads();                   // prior subspace done reading hS/sS
    for (int i = tid; i < kMT * kDS; i += 256) {
      const int row = i / kDS, dl = i % kDS;
      const int d = n * kDS + dl;
      const int dp = (d < 512) ? d + 512 : d - 512;   // rotate_half partner
      const int t = t0 + row;
      const float* xr = x + (size_t)t * kDn;
      const float rn = rmsS[row];
      const float hd = xr[d] * rn * norm_w[d];
      const float hp = xr[dp] * rn * norm_w[dp];
      const int j = d & 511;           // freq index (emb = concat[f, f])
      const float fr = __expf((float)j * (-9.210340371976184f / 512.f));
      const float th = (float)(t & (kSn - 1)) * fr;
      const float cs = __cosf(th), sn = __sinf(th);
      hS[row][dl] = (bf16)(hd * cs + ((d < 512) ? -hp : hp) * sn);
    }
    __syncthreads();

    // ---- gate/up GEMMs: 32 x 512, K=256; wave owns 2M x 4N tiles ----
    v8f gacc[2][4] = {}, uacc[2][4] = {};
    for (int kb = 0; kb < kDS; kb += 32) {
      const bf16* a0p = &hS[rowA][kb];
      const bf16* a1p = &hS[16 + rowA][kb];
      v16bf a0 = load_a2(a0p + hiA * 16, a0p + hiA * 16 + 8);
      v16bf a1 = load_a2(a1p + hiA * 16, a1p + hiA * 16 + 8);
      #pragma unroll
      for (int tI = 0; tI < 4; ++tI) {
        const int col = (wave * 4 + tI) * 16 + colB;
        v16bf bg = load_b16(WgT + (((size_t)n * kHn + col) * kDS + kb + kHi));
        v16bf bu = load_b16(WuT + (((size_t)n * kHn + col) * kDS + kb + kHi));
        gacc[0][tI] = wmma_bf16(a0, bg, gacc[0][tI]);
        gacc[1][tI] = wmma_bf16(a1, bg, gacc[1][tI]);
        uacc[0][tI] = wmma_bf16(a0, bu, uacc[0][tI]);
        uacc[1][tI] = wmma_bf16(a1, bu, uacc[1][tI]);
      }
    }
    // ---- s = silu(gate) * up (f32) -> LDS bf16 ----
    #pragma unroll
    for (int m = 0; m < 2; ++m)
      #pragma unroll
      for (int tI = 0; tI < 4; ++tI) {
        const int col = (wave * 4 + tI) * 16 + colB;
        #pragma unroll
        for (int v = 0; v < 8; ++v) {
          const float g = gacc[m][tI][v], u = uacc[m][tI][v];
          sS[m * 16 + v + 8 * hiA][col] = (bf16)((g / (1.f + __expf(-g))) * u);
        }
      }
    __syncthreads();

    // ---- down GEMM: 32 x 256, K=512; wave owns 2M x 2N tiles ----
    v8f dacc[2][2] = {};
    for (int kb = 0; kb < kHn; kb += 32) {
      const bf16* a0p = &sS[rowA][kb];
      const bf16* a1p = &sS[16 + rowA][kb];
      v16bf a0 = load_a2(a0p + hiA * 16, a0p + hiA * 16 + 8);
      v16bf a1 = load_a2(a1p + hiA * 16, a1p + hiA * 16 + 8);
      #pragma unroll
      for (int tI = 0; tI < 2; ++tI) {
        const int col = (wave * 2 + tI) * 16 + colB;
        v16bf bd = load_b16(WdT + (((size_t)n * kDS + col) * kHn + kb + kHi));
        dacc[0][tI] = wmma_bf16(a0, bd, dacc[0][tI]);
        dacc[1][tI] = wmma_bf16(a1, bd, dacc[1][tI]);
      }
    }
    #pragma unroll
    for (int m = 0; m < 2; ++m)
      #pragma unroll
      for (int tI = 0; tI < 2; ++tI) {
        const int col = (wave * 2 + tI) * 16 + colB;
        #pragma unroll
        for (int v = 0; v < 8; ++v) {
          const int r = m * 16 + v + 8 * hiA;
          dn[(size_t)(t0 + r) * kDn + n * kDS + col] = (bf16)dacc[m][tI][v];
        }
      }
  }
}

// ---------------------------------------------------------------------------
// out = dn @ Wo + x.  One block = 32 tokens; wave owns 2M x 8N tiles;
// A staged in K-chunks of 256 to keep LDS small; each Wo fragment feeds
// 2 WMMAs.
// ---------------------------------------------------------------------------
__global__ __launch_bounds__(256) void out_proj(
    const bf16* __restrict__ dn, const bf16* __restrict__ WoT,
    const float* __restrict__ x, float* __restrict__ out) {
  __shared__ __align__(16) bf16 aS[kMT][256 + 8];
  const int tid  = threadIdx.x;
  const int lane = tid & 31;
  const int wave = tid >> 5;
  const int t0   = blockIdx.x * kMT;
  const int rowA = lane & 15, hiA = lane >> 4;
  const int colB = lane & 15, kHi = hiA << 4;

  v8f acc[2][8] = {};
  for (int kc = 0; kc < kDn; kc += 256) {
    __syncthreads();                   // previous chunk fully consumed
    for (int i = tid; i < kMT * 256 / 8; i += 256) {
      const int row = (i * 8) / 256, col = (i * 8) % 256;
      *(uint4*)&aS[row][col] =
          *(const uint4*)(dn + (size_t)(t0 + row) * kDn + kc + col);
    }
    __syncthreads();
    for (int kb = 0; kb < 256; kb += 32) {
      const bf16* a0p = &aS[rowA][kb];
      const bf16* a1p = &aS[16 + rowA][kb];
      v16bf a0 = load_a2(a0p + hiA * 16, a0p + hiA * 16 + 8);
      v16bf a1 = load_a2(a1p + hiA * 16, a1p + hiA * 16 + 8);
      #pragma unroll
      for (int tI = 0; tI < 8; ++tI) {
        const int col = wave * 128 + tI * 16 + colB;
        v16bf b = load_b16(WoT + ((size_t)col * kDn + kc + kb + kHi));
        acc[0][tI] = wmma_bf16(a0, b, acc[0][tI]);
        acc[1][tI] = wmma_bf16(a1, b, acc[1][tI]);
      }
    }
  }
  #pragma unroll
  for (int m = 0; m < 2; ++m)
    #pragma unroll
    for (int tI = 0; tI < 8; ++tI) {
      const int col = wave * 128 + tI * 16 + colB;
      #pragma unroll
      for (int v = 0; v < 8; ++v) {
        const int t = t0 + m * 16 + v + 8 * hiA;
        out[(size_t)t * kDn + col] = acc[m][tI][v] + x[(size_t)t * kDn + col];
      }
    }
}

// ---------------------------------------------------------------------------
extern "C" void kernel_launch(void* const* d_in, const int* in_sizes, int n_in,
                              void* d_out, int out_size, void* d_ws,
                              size_t ws_size, hipStream_t stream) {
  (void)in_sizes; (void)n_in; (void)out_size; (void)ws_size;
  const float* x      = (const float*)d_in[0];
  const float* norm_w = (const float*)d_in[1];
  const float* Wg     = (const float*)d_in[2];
  const float* Wu     = (const float*)d_in[3];
  const float* Wd     = (const float*)d_in[4];
  const float* Wo     = (const float*)d_in[5];
  float* out = (float*)d_out;
  char*  ws  = (char*)d_ws;

  bf16* dn  = (bf16*)(ws + kDnOff);
  bf16* WgT = (bf16*)(ws + kWgTOff);
  bf16* WuT = (bf16*)(ws + kWuTOff);
  bf16* WdT = (bf16*)(ws + kWdTOff);
  bf16* WoT = (bf16*)(ws + kWoTOff);

  // Weight convert + transpose (bf16, K contiguous for B operand)
  for (int n = 0; n < kNS; ++n) {
    transpose_cvt<<<(kDS * kHn + 255) / 256, 256, 0, stream>>>(
        Wg + (size_t)n * kDS * kHn, WgT + (size_t)n * kHn * kDS, kDS, kHn);
    transpose_cvt<<<(kDS * kHn + 255) / 256, 256, 0, stream>>>(
        Wu + (size_t)n * kDS * kHn, WuT + (size_t)n * kHn * kDS, kDS, kHn);
    transpose_cvt<<<(kHn * kDS + 255) / 256, 256, 0, stream>>>(
        Wd + (size_t)n * kHn * kDS, WdT + (size_t)n * kDS * kHn, kHn, kDS);
  }
  transpose_cvt<<<(kDn * kDn + 255) / 256, 256, 0, stream>>>(Wo, WoT, kDn, kDn);

  fused_norm_rope_glu<<<kT / kMT, 256, 0, stream>>>(x, norm_w, dn, WgT, WuT, WdT);
  out_proj<<<kT / kMT, 256, 0, stream>>>(dn, WoT, x, out);
}
